// HGNNLayer_7997229105162
// MI455X (gfx1250) — matrix operand order repaired and checked by reading, not compiled, not run on hardware
//
#include <hip/hip_runtime.h>
#include <hip/hip_bf16.h>

// ---------------------------------------------------------------------------
// Types for CDNA5 WMMA
// ---------------------------------------------------------------------------
typedef __bf16 bf16_t;
typedef bf16_t v16bf __attribute__((ext_vector_type(16)));
typedef float  v8f   __attribute__((ext_vector_type(8)));

struct U8x32 { uint4 a, b; };   // 32 bytes == v16bf

static __device__ __forceinline__ v16bf frag_from(uint4 a, uint4 b) {
    U8x32 u; u.a = a; u.b = b;
    return __builtin_bit_cast(v16bf, u);
}

static __device__ __forceinline__ unsigned short bf_bits(float f) {
    union { float f; unsigned u; } v; v.f = f;
    unsigned u = v.u;
    u += 0x7FFFu + ((u >> 16) & 1u);   // round-to-nearest-even
    return (unsigned short)(u >> 16);
}

static __device__ __forceinline__ unsigned pack_bf2(float lo, float hi) {
    return (unsigned)bf_bits(lo) | ((unsigned)bf_bits(hi) << 16);
}

static __device__ __forceinline__ float lrelu(float x, float s) {
    return x >= 0.0f ? x : s * x;
}

static __device__ __forceinline__ float red16(float v) {
    // sum across the 16-lane half-wave group (C-matrix row group)
    v += __shfl_xor(v, 1);
    v += __shfl_xor(v, 2);
    v += __shfl_xor(v, 4);
    v += __shfl_xor(v, 8);
    return v;
}

static __device__ __forceinline__ void atomicMaxF(float* addr, float val) {
    if (val >= 0.0f) atomicMax((int*)addr, __float_as_int(val));
    else             atomicMin((unsigned int*)addr, __float_as_uint(val));
}

#define D_DIM 128

// ---------------------------------------------------------------------------
// Weight prep: build bf16 B-matrices in workspace.
//  bufs 0..2 : Wt[r][j][k] = W[r][k][j]   (for h = x @ W[r],  B[k][j] = W[k][j])
//  buf  3    : Wt[j][k]    = bi_w[j][k]   (for dfm @ bi_w^T,  B[k][j] = bi_w[j][k])
//  buf  4    : Wt[j][k]    = si_w[j][k]
// Each buffer is 128x128 ushort = 32KB; fragment loads read 16 contiguous bf16.
// ---------------------------------------------------------------------------
__global__ void prep_w_kernel(const float* __restrict__ W,
                              const float* __restrict__ biw,
                              const float* __restrict__ siw,
                              unsigned short* __restrict__ wt) {
    int idx = blockIdx.x * 256 + threadIdx.x;
    if (idx >= 5 * 16384) return;
    int buf = idx >> 14, e = idx & 16383;
    int j = e >> 7, k = e & 127;
    float v;
    if (buf < 3)        v = W[(size_t)buf * 16384 + (size_t)k * 128 + j];
    else if (buf == 3)  v = biw[e];
    else                v = siw[e];
    wt[idx] = bf_bits(v);
}

// ---------------------------------------------------------------------------
// GEMM: h[node0:node0+128, :] = x @ W  via v_wmma_f32_16x16x32_bf16
// Block = 256 threads = 8 waves; wave w owns rows [w*16, w*16+16).
// ---------------------------------------------------------------------------
__global__ __launch_bounds__(256) void gemm_h_kernel(
    const float* __restrict__ x, const unsigned short* __restrict__ wt,
    float* __restrict__ h, int N) {
    __shared__ __align__(16) unsigned int aLds[128 * 64];  // 128 rows x 128 bf16

    const int tid = threadIdx.x;
    const int node0 = blockIdx.x * 128;

    for (int u = tid; u < 128 * 64; u += 256) {
        int row = u >> 6, cp = u & 63;
        int node = node0 + row;
        float2 v = make_float2(0.f, 0.f);
        if (node < N) v = *(const float2*)(x + (size_t)node * D_DIM + 2 * cp);
        aLds[u] = pack_bf2(v.x, v.y);
    }
    __syncthreads();

    const int wv = tid >> 5, lane = tid & 31;
    const int hi = (lane >= 16) ? 1 : 0;
    const int ln16 = lane & 15;
    const int m0 = wv * 16;

    v8f zero = {0.f, 0.f, 0.f, 0.f, 0.f, 0.f, 0.f, 0.f};
    v8f acc[8];
#pragma unroll
    for (int n = 0; n < 8; ++n) acc[n] = zero;

    const uint4* arow = (const uint4*)aLds + (size_t)(m0 + ln16) * 16;
#pragma unroll
    for (int kb = 0; kb < 128; kb += 32) {
        int ca = (kb >> 3) + hi * 2;
        v16bf afrag = frag_from(arow[ca], arow[ca + 1]);
#pragma unroll
        for (int n = 0; n < 8; ++n) {
            const uint4* brow = (const uint4*)(wt + (size_t)(n * 16 + ln16) * D_DIM);
            int cb = (kb >> 3) + hi * 2;
            v16bf bfrag = frag_from(brow[cb], brow[cb + 1]);
            acc[n] = __builtin_amdgcn_wmma_f32_16x16x32_bf16(
                false, afrag, false, bfrag, (short)0, acc[n], false, false);
        }
    }

#pragma unroll
    for (int n = 0; n < 8; ++n) {
        int col = n * 16 + ln16;
#pragma unroll
        for (int r = 0; r < 8; ++r) {
            int node = node0 + m0 + r + hi * 8;
            if (node < N) h[(size_t)node * D_DIM + col] = acc[n][r];
        }
    }
}

// ---------------------------------------------------------------------------
// Per-node attention scores: el = <h, al>, er = <h, ar>.  One wave per node.
// ---------------------------------------------------------------------------
__global__ void attn_kernel(const float* __restrict__ h,
                            const float* __restrict__ al,
                            const float* __restrict__ ar,
                            float* __restrict__ el, float* __restrict__ er, int N) {
    int wv = threadIdx.x >> 5, lane = threadIdx.x & 31;
    int node = blockIdx.x * 8 + wv;
    if (node >= N) return;
    const float* hp = h + (size_t)node * D_DIM;
    float sl = 0.f, sr = 0.f;
#pragma unroll
    for (int i = 0; i < 4; ++i) {
        float v = hp[lane + 32 * i];
        sl += v * al[lane + 32 * i];
        sr += v * ar[lane + 32 * i];
    }
#pragma unroll
    for (int m = 16; m >= 1; m >>= 1) {
        sl += __shfl_xor(sl, m);
        sr += __shfl_xor(sr, m);
    }
    if (lane == 0) { el[node] = sl; er[node] = sr; }
}

__global__ void init_node_kernel(float* __restrict__ mmax, float* __restrict__ ssum, int N) {
    int i = blockIdx.x * 256 + threadIdx.x;
    if (i < N) { mmax[i] = -__builtin_inff(); ssum[i] = 0.f; }
}

__global__ void init_feats_kernel(float* __restrict__ feats,
                                  const float* __restrict__ gat_b, size_t ND) {
    size_t i = (size_t)blockIdx.x * 256 + threadIdx.x;
    if (i < ND) feats[i] = gat_b[i & (D_DIM - 1)];
}

__global__ void zero2_kernel(float* __restrict__ a, float* __restrict__ b, size_t n) {
    size_t i = (size_t)blockIdx.x * 256 + threadIdx.x;
    if (i < n) { a[i] = 0.f; b[i] = 0.f; }
}

// ---------------------------------------------------------------------------
// Edge softmax pass 1: e = LeakyReLU(el[src]+er[dst], 0.2); segment max via atomics
// ---------------------------------------------------------------------------
__global__ void edge_max_kernel(const int* __restrict__ src, const int* __restrict__ dst,
                                const float* __restrict__ el, const float* __restrict__ er,
                                float* __restrict__ ebuf, float* __restrict__ mmax, int E) {
    int e = blockIdx.x * 256 + threadIdx.x;
    if (e >= E) return;
    int s = src[e], d = dst[e];
    float v = lrelu(el[s] + er[d], 0.2f);
    ebuf[e] = v;
    atomicMaxF(&mmax[d], v);
}

// Pass 2: a = exp(e - m[dst]); segment sum
__global__ void edge_exp_kernel(const float* __restrict__ ebuf, const int* __restrict__ dst,
                                const float* __restrict__ mmax, float* __restrict__ abuf,
                                float* __restrict__ ssum, int E) {
    int e = blockIdx.x * 256 + threadIdx.x;
    if (e >= E) return;
    int d = dst[e];
    float a = __expf(ebuf[e] - mmax[d]);
    abuf[e] = a;
    atomicAdd(&ssum[d], a);
}

// Pass 3: alpha = a/s[dst]*ew; feats[dst] += alpha * h[src].  One wave per edge.
__global__ __launch_bounds__(256) void edge_scatter_kernel(
    const int* __restrict__ src, const int* __restrict__ dst,
    const float* __restrict__ ew, const float* __restrict__ abuf,
    const float* __restrict__ ssum, const float* __restrict__ h,
    float* __restrict__ feats, int E) {
    int wv = threadIdx.x >> 5, lane = threadIdx.x & 31;
    int e = blockIdx.x * 8 + wv;
    if (e >= E) return;
    int s = src[e], d = dst[e];
    float alpha = abuf[e] / ssum[d] * ew[e];
    float4 v = ((const float4*)(h + (size_t)s * D_DIM))[lane];
    float* fp = feats + (size_t)d * D_DIM + lane * 4;
    atomicAdd(fp + 0, alpha * v.x);
    atomicAdd(fp + 1, alpha * v.y);
    atomicAdd(fp + 2, alpha * v.z);
    atomicAdd(fp + 3, alpha * v.w);
}

// Accumulate FM statistics: s += feats; sq += feats^2
__global__ void accum_kernel(const float* __restrict__ feats,
                             float* __restrict__ s, float* __restrict__ sq, size_t n) {
    size_t i = (size_t)blockIdx.x * 256 + threadIdx.x;
    if (i < n) { float f = feats[i]; s[i] += f; sq[i] += f * f; }
}

// ---------------------------------------------------------------------------
// FM bi-interaction + two WMMA GEMMs + gated residual + LayerNorm, fused.
// ---------------------------------------------------------------------------
__global__ __launch_bounds__(256) void fm_ln_kernel(
    const float* __restrict__ s, const float* __restrict__ sq,
    const float* __restrict__ x,
    const unsigned short* __restrict__ wtb, const float* __restrict__ bib,
    const unsigned short* __restrict__ wts, const float* __restrict__ sib,
    const float* __restrict__ resw, const float* __restrict__ lng,
    const float* __restrict__ lnb, float* __restrict__ out, int N) {
    __shared__ __align__(16) unsigned int a1Lds[128 * 64];  // deep_fm (bf16)
    __shared__ __align__(16) unsigned int a2Lds[128 * 64];  // s       (bf16)

    const int tid = threadIdx.x;
    const int node0 = blockIdx.x * 128;

    for (int u = tid; u < 128 * 64; u += 256) {
        int row = u >> 6, cp = u & 63;
        int node = node0 + row;
        float2 sv = make_float2(0.f, 0.f), qv = make_float2(0.f, 0.f);
        if (node < N) {
            sv = *(const float2*)(s  + (size_t)node * D_DIM + 2 * cp);
            qv = *(const float2*)(sq + (size_t)node * D_DIM + 2 * cp);
        }
        float d0 = 0.5f * (sv.x * sv.x - qv.x);
        float d1 = 0.5f * (sv.y * sv.y - qv.y);
        a1Lds[u] = pack_bf2(d0, d1);
        a2Lds[u] = pack_bf2(sv.x, sv.y);
    }
    __syncthreads();

    const int wv = tid >> 5, lane = tid & 31;
    const int hi = (lane >= 16) ? 1 : 0;
    const int ln16 = lane & 15;
    const int m0 = wv * 16;

    v8f zero = {0.f, 0.f, 0.f, 0.f, 0.f, 0.f, 0.f, 0.f};
    v8f acc1[8], acc2[8];
#pragma unroll
    for (int n = 0; n < 8; ++n) { acc1[n] = zero; acc2[n] = zero; }

    const uint4* a1row = (const uint4*)a1Lds + (size_t)(m0 + ln16) * 16;
    const uint4* a2row = (const uint4*)a2Lds + (size_t)(m0 + ln16) * 16;
#pragma unroll
    for (int kb = 0; kb < 128; kb += 32) {
        int ca = (kb >> 3) + hi * 2;
        v16bf af1 = frag_from(a1row[ca], a1row[ca + 1]);
        v16bf af2 = frag_from(a2row[ca], a2row[ca + 1]);
#pragma unroll
        for (int n = 0; n < 8; ++n) {
            int cb = (kb >> 3) + hi * 2;
            const uint4* b1 = (const uint4*)(wtb + (size_t)(n * 16 + ln16) * D_DIM);
            const uint4* b2 = (const uint4*)(wts + (size_t)(n * 16 + ln16) * D_DIM);
            v16bf bf1 = frag_from(b1[cb], b1[cb + 1]);
            v16bf bf2 = frag_from(b2[cb], b2[cb + 1]);
            acc1[n] = __builtin_amdgcn_wmma_f32_16x16x32_bf16(
                false, af1, false, bf1, (short)0, acc1[n], false, false);
            acc2[n] = __builtin_amdgcn_wmma_f32_16x16x32_bf16(
                false, af2, false, bf2, (short)0, acc2[n], false, false);
        }
    }

    const float sig = 1.f / (1.f + __expf(-resw[0]));

#pragma unroll
    for (int r = 0; r < 8; ++r) {
        int node = node0 + m0 + r + hi * 8;
        bool live = node < N;
        float v[8];
        float psum = 0.f;
#pragma unroll
        for (int n = 0; n < 8; ++n) {
            int col = n * 16 + ln16;
            float xv = live ? x[(size_t)node * D_DIM + col] : 0.f;
            float t = lrelu(acc1[n][r] + bib[col], 0.01f)
                    + lrelu(acc2[n][r] + sib[col], 0.01f)
                    + xv * sig;
            v[n] = t;
            psum += t;
        }
        float mu = red16(psum) * (1.f / 128.f);
        float pvar = 0.f;
#pragma unroll
        for (int n = 0; n < 8; ++n) { float dd = v[n] - mu; pvar += dd * dd; }
        float inv = rsqrtf(red16(pvar) * (1.f / 128.f) + 1e-5f);
        if (live) {
#pragma unroll
            for (int n = 0; n < 8; ++n) {
                int col = n * 16 + ln16;
                out[(size_t)node * D_DIM + col] = (v[n] - mu) * inv * lng[col] + lnb[col];
            }
        }
    }
}

// ---------------------------------------------------------------------------
// Launch
// ---------------------------------------------------------------------------
extern "C" void kernel_launch(void* const* d_in, const int* in_sizes, int n_in,
                              void* d_out, int out_size, void* d_ws, size_t ws_size,
                              hipStream_t stream) {
    const float* x        = (const float*)d_in[0];
    const int*   edge_src = (const int*)  d_in[1];
    const int*   edge_dst = (const int*)  d_in[2];
    const float* edge_w   = (const float*)d_in[3];
    const float* W        = (const float*)d_in[4];
    const float* attn_l   = (const float*)d_in[5];
    const float* attn_r   = (const float*)d_in[6];
    const float* gat_b    = (const float*)d_in[7];
    const float* bi_w     = (const float*)d_in[8];
    const float* bi_b     = (const float*)d_in[9];
    const float* si_w     = (const float*)d_in[10];
    const float* si_b     = (const float*)d_in[11];
    const float* res_w    = (const float*)d_in[12];
    const float* ln_g     = (const float*)d_in[13];
    const float* ln_b     = (const float*)d_in[14];
    float* out = (float*)d_out;

    const int D = D_DIM;
    const int N = in_sizes[0] / D;
    const int R = in_sizes[4] / (D * D);
    const int E = in_sizes[1] / R;
    const size_t ND = (size_t)N * D;

    // workspace carve-out
    float* f = (float*)d_ws;
    float* h     = f; f += ND;
    float* feats = f; f += ND;
    float* sbuf  = f; f += ND;
    float* sqbuf = f; f += ND;
    float* el    = f; f += N;
    float* er    = f; f += N;
    float* mmax  = f; f += N;
    float* ssum  = f; f += N;
    float* ebuf  = f; f += E;
    float* abuf  = f; f += E;
    unsigned short* wt = (unsigned short*)f;  // 5 * 16384 ushort (bf16 weights)

    const int tilesM   = (N + 127) / 128;
    const int ndBlocks = (int)((ND + 255) / 256);

    zero2_kernel<<<ndBlocks, 256, 0, stream>>>(sbuf, sqbuf, ND);
    prep_w_kernel<<<(5 * 16384 + 255) / 256, 256, 0, stream>>>(W, bi_w, si_w, wt);

    for (int r = 0; r < R; ++r) {
        const int*   src = edge_src + (size_t)r * E;
        const int*   dst = edge_dst + (size_t)r * E;
        const float* ew  = edge_w   + (size_t)r * E;

        gemm_h_kernel<<<tilesM, 256, 0, stream>>>(x, wt + (size_t)r * 16384, h, N);
        attn_kernel<<<(N + 7) / 8, 256, 0, stream>>>(h, attn_l + (size_t)r * D,
                                                     attn_r + (size_t)r * D, el, er, N);
        init_node_kernel<<<(N + 255) / 256, 256, 0, stream>>>(mmax, ssum, N);
        init_feats_kernel<<<ndBlocks, 256, 0, stream>>>(feats, gat_b + (size_t)r * D, ND);
        edge_max_kernel<<<(E + 255) / 256, 256, 0, stream>>>(src, dst, el, er, ebuf, mmax, E);
        edge_exp_kernel<<<(E + 255) / 256, 256, 0, stream>>>(ebuf, dst, mmax, abuf, ssum, E);
        edge_scatter_kernel<<<(E + 7) / 8, 256, 0, stream>>>(src, dst, ew, abuf, ssum, h, feats, E);
        accum_kernel<<<ndBlocks, 256, 0, stream>>>(feats, sbuf, sqbuf, ND);
    }

    fm_ln_kernel<<<tilesM, 256, 0, stream>>>(sbuf, sqbuf, x, wt + 3 * 16384, bi_b,
                                             wt + 4 * 16384, si_b, res_w, ln_g, ln_b, out, N);
}
